// tidigitsRNNPT_65927747993823
// MI455X (gfx1250) — matrix operand order repaired and checked by reading, not compiled, not run on hardware
//
#include <hip/hip_runtime.h>
#include <hip/hip_bf16.h>
#include <math.h>

#define FEAT 39
#define HID 1024
#define SEQ 254
#define NCLS 10
#define FC1N 100
#define FC1P 112   // FC1 padded to multiple of 16
#define BATCH 128

typedef __bf16 bf16;
typedef __attribute__((ext_vector_type(8)))  __bf16 v8bf;
typedef __attribute__((ext_vector_type(16))) __bf16 v16bf;
typedef __attribute__((ext_vector_type(8)))  float  v8f;
typedef __attribute__((ext_vector_type(4)))  unsigned v4u;
typedef __attribute__((ext_vector_type(8)))  unsigned v8u;

static __device__ __forceinline__ v16bf cat16(v8bf lo, v8bf hi) {
  return __builtin_shufflevector(lo, hi, 0,1,2,3,4,5,6,7,8,9,10,11,12,13,14,15);
}

// Branch-free tanh. Prefer the CDNA5 hardware v_tanh_f32 if exposed.
static __device__ __forceinline__ float fast_tanh(float x) {
#if __has_builtin(__builtin_amdgcn_tanhf)
  return __builtin_amdgcn_tanhf(x);
#else
  float cx = fminf(fmaxf(x, -12.0f), 12.0f);
  float e  = __expf(2.0f * cx);                       // v_exp_f32
  return (e - 1.0f) * __builtin_amdgcn_rcpf(e + 1.0f); // raw v_rcp_f32
#endif
}

// LDS byte offset of a (dynamic) shared-memory pointer, via addrspacecast.
typedef __attribute__((address_space(3))) char lds_char;
static __device__ __forceinline__ unsigned lds_off_of(void* p) {
  return (unsigned)(unsigned long long)(lds_char*)p;
}

// CDNA5 async memory->LDS copy, 16 bytes per lane per issue (ASYNCcnt).
static __device__ __forceinline__ void async_copy_b128(unsigned lds_byte_addr,
                                                       const void* gaddr) {
  asm volatile("global_load_async_to_lds_b128 %0, %1, off"
               :: "v"(lds_byte_addr), "v"(gaddr) : "memory");
}
static __device__ __forceinline__ void async_wait0() {
  asm volatile("s_wait_asynccnt 0" ::: "memory");
}

// ---------------------------------------------------------------------------
// Tensor Data Mover: 1-D contiguous tile, global -> LDS (TENSORcnt).
// D# per CDNA5 ISA: data_size=3 (8B elems), count=1,
// type=2; tile_dim0 = elems8b (<= 65535); groups 2/3 = NULL (<=2D tensor).
// Issue from ONE wave only; pair with s_wait_tensorcnt + workgroup barrier.
// ---------------------------------------------------------------------------
static __device__ __forceinline__ void tdm_load_lds(unsigned lds_byte_addr,
                                                    const void* gaddr,
                                                    unsigned elems8b) {
  unsigned long long ga = (unsigned long long)gaddr;
  v4u g0;
  g0[0] = 1u;                                   // count=1, is_restore=0
  g0[1] = lds_byte_addr;                        // lds_addr
  g0[2] = (unsigned)ga;                         // global_addr[31:0]
  g0[3] = (unsigned)((ga >> 32) & 0x01FFFFFFu)  // global_addr[56:32]
          | 0x80000000u;                        // type=2 ("image") in [127:126]
  v8u g1;
  g1[0] = 0x00030000u;                          // data_size=3 (8B), mask=0
  g1[1] = (elems8b & 0xFFFFu) << 16;            // tensor_dim0[15:0]
  g1[2] = (elems8b >> 16) & 0xFFFFu;            // tensor_dim0[31:16]
  g1[3] = (elems8b & 0xFFFFu) << 16;            // tile_dim0
  g1[4] = 0u;                                   // tile_dim1/2 = 0 (unused)
  g1[5] = elems8b;                              // tensor_dim0_stride[31:0]
  g1[6] = 0u;
  g1[7] = 0u;
  asm volatile("tensor_load_to_lds %0, %1" :: "s"(g0), "s"(g1) : "memory");
}

// ---------------------------------------------------------------------------
// Prep: convert W_hh and h0 to bf16 scratch copies
// ---------------------------------------------------------------------------
__global__ void prep_cvt_kernel(const float* __restrict__ W_hh,
                                const float* __restrict__ h0,
                                bf16* __restrict__ whh_bf,
                                bf16* __restrict__ h0_bf) {
  int i = blockIdx.x * blockDim.x + threadIdx.x;
  if (i < HID * HID)   whh_bf[i] = (bf16)W_hh[i];
  if (i < BATCH * HID) h0_bf[i]  = (bf16)h0[i];
}

// ---------------------------------------------------------------------------
// xproj[s][b][h] = bf16( x[b,s,:] . W_ih[h,:] + b_ih[h] + b_hh[h] )
// ---------------------------------------------------------------------------
__global__ void xproj_kernel(const float* __restrict__ x,
                             const float* __restrict__ W_ih,
                             const float* __restrict__ b_ih,
                             const float* __restrict__ b_hh,
                             bf16* __restrict__ xproj) {
  long idx = (long)blockIdx.x * blockDim.x + threadIdx.x;
  const long total = (long)SEQ * BATCH * HID;
  if (idx >= total) return;
  int h = (int)(idx & (HID - 1));
  int b = (int)((idx >> 10) & (BATCH - 1));
  int s = (int)(idx >> 17);                 // BATCH*HID = 2^17
  const float* xr = x + ((long)b * SEQ + s) * FEAT;
  const float* wr = W_ih + (long)h * FEAT;
  float sum = b_ih[h] + b_hh[h];
#pragma unroll
  for (int f = 0; f < FEAT; ++f) sum += xr[f] * wr[f];
  xproj[idx] = (bf16)sum;
}

// ---------------------------------------------------------------------------
// Monotonic grid barrier (64 persistent workgroups, one phase per timestep)
// ---------------------------------------------------------------------------
__device__ __forceinline__ void grid_barrier(unsigned* cnt, unsigned nwg) {
  __syncthreads();
  if (threadIdx.x == 0) {
    __threadfence();
    unsigned arrive = atomicAdd(cnt, 1u) + 1u;
    unsigned target = ((arrive - 1u) / nwg + 1u) * nwg;
    while (atomicAdd(cnt, 0u) < target) __builtin_amdgcn_s_sleep(1);
  }
  __syncthreads();
}

// ---------------------------------------------------------------------------
// Persistent recurrent scan.
// 64 WGs = 8 batch-tiles(16) x 8 hidden-slices(128). W_hh slice (256KB)
// TDM-loaded once and resident in LDS for all 254 steps; per-step h tile
// staged with async global->LDS; 32x v_wmma_f32_16x16x32_bf16 per wave/step.
// ---------------------------------------------------------------------------
__global__ void __launch_bounds__(256, 1)
rnn_scan_kernel(const bf16* __restrict__ whh_bf,
                const bf16* __restrict__ h0_bf,
                bf16* __restrict__ xo,      // [SEQ][BATCH][HID] xproj in / outs out
                unsigned* cnt) {
  extern __shared__ char smem[];
  bf16* ldsB = (bf16*)smem;                              // 128*1024 bf16
  bf16* ldsA = (bf16*)(smem + 128 * HID * sizeof(bf16)); // 16*1024 bf16

  const int tid   = threadIdx.x;
  const int lane  = tid & 31;
  const int wave  = tid >> 5;
  const int wg_m  = blockIdx.x & 7;     // batch tile
  const int wg_n  = blockIdx.x >> 3;    // hidden slice
  const int m0    = wg_m * 16;
  const int nbase = wg_n * 128;

  const unsigned ldsB_base = lds_off_of(ldsB);
  const unsigned ldsA_base = lds_off_of(ldsA);

  // One-time: TDM DMA of W_hh rows [nbase, nbase+128) into LDS (256 KB).
  if (wave == 0) {
    tdm_load_lds(ldsB_base, whh_bf + (long)nbase * HID, 128 * HID * 2 / 8);
    __builtin_amdgcn_s_wait_tensorcnt(0);
  }

  const int lhalf = lane >> 4;   // 0/1: which K-half this lane holds
  const int l16   = lane & 15;
  const int nsub  = wave * 16;   // local col base within slice

  for (int s = 0; s < SEQ; ++s) {
    // Stage A: rows m0..m0+15 of h_s (h0 for s=0, else outs[s-1])
    const char* hsrc = (s == 0) ? (const char*)(h0_bf + (long)m0 * HID)
                                : (const char*)(xo + ((long)(s - 1) * BATCH + m0) * HID);
    for (int i = tid; i < 16 * HID * 2 / 16; i += 256)
      async_copy_b128(ldsA_base + i * 16u, hsrc + (size_t)i * 16);
    async_wait0();
    __syncthreads();   // also orders wave0's TDM completion with all waves

    // Accumulator init from xproj[s] (C/D layout: VGPR r, lane -> M=r+8*lhalf, N=l16)
    v8f acc;
    const bf16* xp = xo + (long)s * BATCH * HID;
#pragma unroll
    for (int r = 0; r < 8; ++r) {
      int m = m0 + r + 8 * lhalf;
      acc[r] = (float)xp[(long)m * HID + nbase + nsub + l16];
    }

    // K loop: 1024 = 32 x K32 WMMAs
#pragma unroll 4
    for (int kk = 0; kk < HID / 32; ++kk) {
      int ka = kk * 32 + lhalf * 16;  // A: lane -> row, 16 contiguous K
      v8bf a_lo = *(const v8bf*)(ldsA + l16 * HID + ka);
      v8bf a_hi = *(const v8bf*)(ldsA + l16 * HID + ka + 8);
      v16bf a = cat16(a_lo, a_hi);
      int kb = kk * 32 + lhalf * 16;  // B: lane -> col, 16 contiguous K
      v8bf b_lo = *(const v8bf*)(ldsB + (nsub + l16) * HID + kb);
      v8bf b_hi = *(const v8bf*)(ldsB + (nsub + l16) * HID + kb + 8);
      v16bf b = cat16(b_lo, b_hi);
      acc = __builtin_amdgcn_wmma_f32_16x16x32_bf16(false, a, false, b,
                                                    (short)0, acc, false, false);
    }

    // tanh + store in place (becomes h_{s+1} and the FC1 feature row)
    bf16* outp = xo + (long)s * BATCH * HID;
#pragma unroll
    for (int r = 0; r < 8; ++r) {
      int m = m0 + r + 8 * lhalf;
      outp[(long)m * HID + nbase + nsub + l16] = (bf16)fast_tanh(acc[r]);
    }
    __threadfence();
    grid_barrier(cnt, 64);
  }
}

// ---------------------------------------------------------------------------
// FC1: h1[128,100] += feats . W1^T over K-chunk s (K=1024 per chunk).
// One WG per timestep; TDM-stage feats slab (256KB) into LDS, stream W1
// from HBM exactly once (with prefetch), WMMA-accumulate, atomicAdd fp32.
// ---------------------------------------------------------------------------
__global__ void __launch_bounds__(256, 1)
fc1_kernel(const bf16* __restrict__ outs,
           const float* __restrict__ W1,
           float* __restrict__ h1) {
  extern __shared__ char smem2[];
  bf16* ldsA = (bf16*)smem2;   // 128*1024 bf16
  const int s    = blockIdx.x;
  const int tid  = threadIdx.x;
  const int lane = tid & 31;
  const int wave = tid >> 5;

  if (wave == 0) {
    tdm_load_lds(lds_off_of(ldsA), outs + (long)s * BATCH * HID,
                 BATCH * HID * 2 / 8);
    __builtin_amdgcn_s_wait_tensorcnt(0);
  }
  __syncthreads();
  if (wave >= 7) return;   // 7 N-tiles of 16 cover FC1P=112

  const int lhalf = lane >> 4;
  const int l16   = lane & 15;
  const int n     = wave * 16 + l16;   // output column (0..111)
  const bool nvalid = n < FC1N;

  v8f acc[8] = {};
  const float* wrow = W1 + (long)n * ((long)SEQ * HID) + (long)s * HID;

  for (int kk = 0; kk < HID / 32; ++kk) {
    int kb = kk * 32 + lhalf * 16;
    v16bf bfrag;
    if (nvalid) {
      __builtin_prefetch(wrow + kb + 128, 0, 1);   // global_prefetch_b8
      const float4* wp = (const float4*)(wrow + kb);
#pragma unroll
      for (int q = 0; q < 4; ++q) {
        float4 f = wp[q];
        bfrag[q * 4 + 0] = (bf16)f.x; bfrag[q * 4 + 1] = (bf16)f.y;
        bfrag[q * 4 + 2] = (bf16)f.z; bfrag[q * 4 + 3] = (bf16)f.w;
      }
    } else {
#pragma unroll
      for (int q = 0; q < 16; ++q) bfrag[q] = (bf16)0.0f;
    }
#pragma unroll
    for (int mt = 0; mt < 8; ++mt) {
      int ka = kk * 32 + lhalf * 16;
      v8bf alo = *(const v8bf*)(ldsA + (mt * 16 + l16) * HID + ka);
      v8bf ahi = *(const v8bf*)(ldsA + (mt * 16 + l16) * HID + ka + 8);
      v16bf a = cat16(alo, ahi);
      acc[mt] = __builtin_amdgcn_wmma_f32_16x16x32_bf16(false, a, false, bfrag,
                                                        (short)0, acc[mt], false, false);
    }
  }
#pragma unroll
  for (int mt = 0; mt < 8; ++mt) {
#pragma unroll
    for (int r = 0; r < 8; ++r) {
      int m = mt * 16 + r + 8 * lhalf;
      atomicAdd(&h1[m * FC1P + n], acc[mt][r]);
    }
  }
}

// ---------------------------------------------------------------------------
// Head: relu(h1+b1) -> logits -> log_softmax. One block per batch row.
// ---------------------------------------------------------------------------
__global__ void head_kernel(const float* __restrict__ h1,
                            const float* __restrict__ b1,
                            const float* __restrict__ W2,
                            const float* __restrict__ b2,
                            float* __restrict__ out) {
  __shared__ float r[FC1N];
  __shared__ float lg[NCLS];
  const int b = blockIdx.x, t = threadIdx.x;
  if (t < FC1N) {
    float v = h1[b * FC1P + t] + b1[t];
    r[t] = v > 0.0f ? v : 0.0f;
  }
  __syncthreads();
  if (t < NCLS) {
    float acc = b2[t];
    for (int j = 0; j < FC1N; ++j) acc += r[j] * W2[t * FC1N + j];
    lg[t] = acc;
  }
  __syncthreads();
  if (t == 0) {
    float m = lg[0];
    for (int c = 1; c < NCLS; ++c) m = fmaxf(m, lg[c]);
    float sum = 0.0f;
    for (int c = 0; c < NCLS; ++c) sum += __expf(lg[c] - m);
    float lse = m + __logf(sum);
    for (int c = 0; c < NCLS; ++c) out[b * NCLS + c] = lg[c] - lse;
  }
}

// ---------------------------------------------------------------------------
extern "C" void kernel_launch(void* const* d_in, const int* in_sizes, int n_in,
                              void* d_out, int out_size, void* d_ws, size_t ws_size,
                              hipStream_t stream) {
  const float* x    = (const float*)d_in[0];
  const float* h0   = (const float*)d_in[1];
  const float* W_ih = (const float*)d_in[2];
  const float* W_hh = (const float*)d_in[3];
  const float* b_ih = (const float*)d_in[4];
  const float* b_hh = (const float*)d_in[5];
  const float* W1   = (const float*)d_in[6];
  const float* b1   = (const float*)d_in[7];
  const float* W2   = (const float*)d_in[8];
  const float* b2   = (const float*)d_in[9];
  float* out = (float*)d_out;

  char* ws = (char*)d_ws;
  size_t off = 0;
  bf16* xo   = (bf16*)(ws + off); off += (size_t)SEQ * BATCH * HID * sizeof(bf16); // 66.6 MB
  bf16* whhb = (bf16*)(ws + off); off += (size_t)HID * HID * sizeof(bf16);         // 2 MB
  bf16* h0b  = (bf16*)(ws + off); off += (size_t)BATCH * HID * sizeof(bf16);       // 256 KB
  float* h1  = (float*)(ws + off); off += (size_t)BATCH * FC1P * sizeof(float);    // 56 KB
  unsigned* cnt = (unsigned*)(ws + off); off += 256;

  hipMemsetAsync(cnt, 0, 256, stream);
  hipMemsetAsync(h1, 0, (size_t)BATCH * FC1P * sizeof(float), stream);

  prep_cvt_kernel<<<(HID * HID + 255) / 256, 256, 0, stream>>>(W_hh, h0, whhb, h0b);

  long total = (long)SEQ * BATCH * HID;
  xproj_kernel<<<(int)((total + 255) / 256), 256, 0, stream>>>(x, W_ih, b_ih, b_hh, xo);

  rnn_scan_kernel<<<64, 256, (128 * HID + 16 * HID) * sizeof(bf16), stream>>>(whhb, h0b, xo, cnt);

  fc1_kernel<<<SEQ, 256, BATCH * HID * sizeof(bf16), stream>>>(xo, W1, h1);

  head_kernel<<<BATCH, 128, 0, stream>>>(h1, b1, W2, b2, out);
}
